// NeuralMemory_65764539236890
// MI455X (gfx1250) — matrix operand order repaired and checked by reading, not compiled, not run on hardware
//
#include <hip/hip_runtime.h>
#include <hip/hip_bf16.h>
#include <math.h>

// ---------------- problem constants (from reference) ----------------
#define B_   2
#define S_   512
#define D_   1024
#define H_   16
#define HD_  64
#define ST_  2048
#define LT_  6144
#define KST_ 204        // int(0.1 * 2048)
#define KLT_ 614        // int(0.1 * 6144)
#define NB_  128        // histogram bins for approximate top-k threshold
#define CH_  32         // keys per staged chunk

typedef __attribute__((ext_vector_type(16))) __bf16 v16bf;
typedef __attribute__((ext_vector_type(8)))  float  v8f;
typedef __attribute__((ext_vector_type(4)))  unsigned int u32x4;
typedef __attribute__((ext_vector_type(8)))  int i32x8;
typedef __attribute__((ext_vector_type(4)))  int i32x4;

// ---------------- WMMA helpers ----------------
static __device__ inline v8f wmma_bf16(v16bf a, v16bf b, v8f c) {
    return __builtin_amdgcn_wmma_f32_16x16x32_bf16(
        false, a, false, b, (short)0, c, false, false);
}

// A fragment (16x32 bf16, row-major, element stride `stride`):
// lanes 0-15: row=lane, K 0..15; lanes 16-31: row=lane-16, K 16..31.
static __device__ inline v16bf load_afrag(const __bf16* base, int stride, int lane) {
    int r = lane & 15;
    int kbase = (lane < 16) ? 0 : 16;
    const __bf16* p = base + (size_t)r * stride + kbase;
    v16bf f;
#pragma unroll
    for (int i = 0; i < 8; ++i) f[i] = p[i];
#pragma unroll
    for (int i = 0; i < 8; ++i) f[8 + i] = p[8 + i];
    return f;
}

// B fragment (32x16 KxN bf16) from N-major storage: lane holds column n=lane&15;
// lanes 0-15 carry K=0..15, lanes 16-31 carry K=16..31 (16 contiguous elements).
static __device__ inline v16bf load_bfrag(const __bf16* base, int stride, int lane) {
    int n = lane & 15;
    int kofs = (lane < 16) ? 0 : 16;
    const __bf16* p = base + (size_t)n * stride + kofs;
    v16bf f;
#pragma unroll
    for (int i = 0; i < 16; ++i) f[i] = p[i];
    return f;
}

// half-wave reductions matching WMMA C layout (reg v: lanes0-15 row v, lanes16-31 row v+8)
static __device__ inline float hsum16(float x) {
#pragma unroll
    for (int m = 1; m <= 8; m <<= 1) x += __shfl_xor(x, m, 32);
    return x;
}
static __device__ inline float hmax16(float x) {
#pragma unroll
    for (int m = 1; m <= 8; m <<= 1) x = fmaxf(x, __shfl_xor(x, m, 32));
    return x;
}
static __device__ inline float hmin16(float x) {
#pragma unroll
    for (int m = 1; m <= 8; m <<= 1) x = fminf(x, __shfl_xor(x, m, 32));
    return x;
}

// ---------------- Tensor Data Mover: 2-D tile -> LDS ----------------
// Builds D# group0/group1 per CDNA5 ISA 8.3/8.4 and issues TENSOR_LOAD_TO_LDS.
// x = fastest dim (contiguous, tile_x elems), y rows with stride_x elements.
static __device__ inline void tdm_load_2d(unsigned lds_addr, const void* gaddr,
                                          unsigned tile_x, unsigned tile_y,
                                          unsigned stride_x) {
    unsigned long long ga = (unsigned long long)(size_t)gaddr;
    u32x4 g0;
    g0[0] = 1u;                                      // count=1, user descriptor
    g0[1] = lds_addr;                                // LDS byte address
    g0[2] = (unsigned)(ga & 0xFFFFFFFFu);            // global_addr[31:0]
    g0[3] = (unsigned)((ga >> 32) & 0x01FFFFFFu) | (2u << 30);  // addr[56:32] | type=2
    i32x8 g1;
    g1[0] = (int)(1u << 16);                         // wg_mask=0, data_size=1 (2B)
    g1[1] = (int)((tile_x & 0xFFFFu) << 16);         // tensor_dim0[15:0] (== tile_x)
    g1[2] = (int)(((tile_x >> 16) & 0xFFFFu) | ((tile_y & 0xFFFFu) << 16)); // dim0 hi | dim1 lo
    g1[3] = (int)(((tile_y >> 16) & 0xFFFFu) | ((tile_x & 0xFFFFu) << 16)); // dim1 hi | tile_dim0
    g1[4] = (int)(tile_y & 0xFFFFu);                 // tile_dim1 | tile_dim2=0
    g1[5] = (int)stride_x;                           // tensor_dim0_stride[31:0]
    g1[6] = 0;                                       // stride0[47:32] | stride1[15:0]
    g1[7] = 0;                                       // stride1[47:16]
    i32x4 gz = {0, 0, 0, 0};
#if __clang_major__ >= 23
    i32x8 gz8 = {0, 0, 0, 0, 0, 0, 0, 0};
    __builtin_amdgcn_tensor_load_to_lds(g0, g1, gz, gz, gz8, 0);
#else
    __builtin_amdgcn_tensor_load_to_lds(g0, g1, gz, gz, 0);
#endif
}

static __device__ inline unsigned lds_off(const void* p) {
    return (unsigned)(size_t)p;   // LDS aperture: addr[31:0] is the LDS byte address
}

// ---------------- conversion kernels ----------------
__global__ __launch_bounds__(256)
void f32_to_bf16_kernel(const float* __restrict__ src, __bf16* __restrict__ dst, int n) {
    int i = blockIdx.x * blockDim.x + threadIdx.x;
    if (i < n) dst[i] = (__bf16)src[i];
}

__global__ __launch_bounds__(256)
void transpose_to_bf16_kernel(const float* __restrict__ src, __bf16* __restrict__ dst,
                              int rows, int cols) {
    int i = blockIdx.x * blockDim.x + threadIdx.x;
    int n = rows * cols;
    if (i < n) {
        int r = i / cols, c = i - r * cols;
        dst[(size_t)c * rows + r] = (__bf16)src[i];
    }
}

// ---------------- Q projection (software-pipelined bf16 WMMA) ----------------
__global__ __launch_bounds__(32)
void qproj_kernel(const __bf16* __restrict__ in_bf, const __bf16* __restrict__ wqT,
                  const float* __restrict__ bq, __bf16* __restrict__ q_bf) {
    const int lane = threadIdx.x;
    const int row0 = blockIdx.x * 16;
    const int n0   = blockIdx.y * 64;
    v8f acc[4] = {};

    v16bf a_c = load_afrag(in_bf + (size_t)row0 * D_, D_, lane);
    v16bf b_c[4];
#pragma unroll
    for (int nt = 0; nt < 4; ++nt)
        b_c[nt] = load_bfrag(wqT + (size_t)(n0 + nt * 16) * D_, D_, lane);

    for (int kc = 32; kc < D_; kc += 32) {
        v16bf a_n = load_afrag(in_bf + (size_t)row0 * D_ + kc, D_, lane);
        v16bf b_n[4];
#pragma unroll
        for (int nt = 0; nt < 4; ++nt)
            b_n[nt] = load_bfrag(wqT + (size_t)(n0 + nt * 16) * D_ + kc, D_, lane);
#pragma unroll
        for (int nt = 0; nt < 4; ++nt)
            acc[nt] = wmma_bf16(a_c, b_c[nt], acc[nt]);
        a_c = a_n;
#pragma unroll
        for (int nt = 0; nt < 4; ++nt) b_c[nt] = b_n[nt];
    }
#pragma unroll
    for (int nt = 0; nt < 4; ++nt)
        acc[nt] = wmma_bf16(a_c, b_c[nt], acc[nt]);

    const int colb  = lane & 15;
    const int rhalf = (lane < 16) ? 0 : 8;
#pragma unroll
    for (int nt = 0; nt < 4; ++nt) {
        int col = n0 + nt * 16 + colb;
        float bias = bq[col];
#pragma unroll
        for (int v = 0; v < 8; ++v)
            q_bf[(size_t)(row0 + rhalf + v) * D_ + col] = (__bf16)(acc[nt][v] + bias);
    }
}

// ---------------- sparse attention ----------------
// 128 threads = 4 waves; each wave owns one 16-query tile; K/V chunks are staged
// into LDS by the Tensor Data Mover (wave 0 issues, TENSORcnt + barriers sync),
// double-buffered so the DMA overlaps the WMMA passes.
static __device__ inline v8f score_tile_lds(v16bf a0, v16bf a1, const __bf16* kb,
                                            int t, int lane) {
    const __bf16* base = kb + t * 16 * HD_;
    v16bf b0 = load_bfrag(base, HD_, lane);
    v16bf b1 = load_bfrag(base + 32, HD_, lane);
    v8f acc = {};
    acc = wmma_bf16(a0, b0, acc);
    return wmma_bf16(a1, b1, acc);
}

__global__ __launch_bounds__(128)
void attention_kernel(const __bf16* __restrict__ q_bf, const __bf16* __restrict__ keys_bf,
                      const __bf16* __restrict__ vT_bf, const float* __restrict__ imp,
                      float* __restrict__ out, int M, int kkeep) {
    __shared__ __align__(128) __bf16 kbuf[2][CH_ * HD_];   // 2 x 4KB key tiles
    __shared__ __align__(128) __bf16 vbuf[2][HD_ * CH_];   // 2 x 4KB value tiles
    __shared__ __bf16 latt[4][16 * 32];                    // per-wave att weights
    __shared__ float rowmax16[4][16];
    __shared__ float rowmin16[4][16];
    __shared__ float thr16[4][16];
    __shared__ unsigned int hist[4][16][NB_];              // 32KB

    const int tid = threadIdx.x;
    const int lane = tid & 31;
    const int wv = tid >> 5;
    int x = blockIdx.x;
    const int qt4 = x & 7;  x >>= 3;     // 8 tile-groups (S_/16/4)
    const int h   = x & 15; x >>= 4;
    const int b   = x;
    const int qt = qt4 * 4 + wv;
    const int q0 = b * S_ + qt * 16;
    const int hofs = h * HD_;
    const int nch = M / CH_;
    const bool issuer = (wv == 0);

    const unsigned ka[2] = { lds_off(&kbuf[0][0]), lds_off(&kbuf[1][0]) };
    const unsigned va[2] = { lds_off(&vbuf[0][0]), lds_off(&vbuf[1][0]) };

    // Q fragments for K=0..63 (reused by all passes)
    v16bf a0 = load_afrag(q_bf + (size_t)q0 * D_ + hofs, D_, lane);
    v16bf a1 = load_afrag(q_bf + (size_t)q0 * D_ + hofs + 32, D_, lane);

    const int  colb = lane & 15;
    const bool lohalf = (lane < 16);

    // ---- pass 1: per-row min/max ----
    float rmx[8], rmn[8];
#pragma unroll
    for (int v = 0; v < 8; ++v) { rmx[v] = -3.0e38f; rmn[v] = 3.0e38f; }

    if (issuer) {
        tdm_load_2d(ka[0], keys_bf + hofs, HD_, CH_, D_);
        tdm_load_2d(ka[1], keys_bf + (size_t)CH_ * D_ + hofs, HD_, CH_, D_);
    }
    for (int c = 0; c < nch; ++c) {
        if (issuer) {
            if (c + 2 <= nch) __builtin_amdgcn_s_wait_tensorcnt(1);
            else              __builtin_amdgcn_s_wait_tensorcnt(0);
        }
        __syncthreads();
        const __bf16* kb = kbuf[c & 1];
#pragma unroll
        for (int t = 0; t < 2; ++t) {
            int m0 = c * CH_ + t * 16;
            v8f s = score_tile_lds(a0, a1, kb, t, lane);
            float iw = imp[m0 + colb] * 0.125f;
#pragma unroll
            for (int v = 0; v < 8; ++v) {
                float sv = s[v] * iw;
                rmx[v] = fmaxf(rmx[v], sv);
                rmn[v] = fminf(rmn[v], sv);
            }
        }
        __syncthreads();
        if (issuer && c + 2 < nch)
            tdm_load_2d(ka[c & 1], keys_bf + (size_t)(c + 2) * CH_ * D_ + hofs, HD_, CH_, D_);
    }
#pragma unroll
    for (int v = 0; v < 8; ++v) { rmx[v] = hmax16(rmx[v]); rmn[v] = hmin16(rmn[v]); }
    if (lane == 0)
#pragma unroll
        for (int v = 0; v < 8; ++v) { rowmax16[wv][v] = rmx[v]; rowmin16[wv][v] = rmn[v]; }
    if (lane == 16)
#pragma unroll
        for (int v = 0; v < 8; ++v) { rowmax16[wv][8 + v] = rmx[v]; rowmin16[wv][8 + v] = rmn[v]; }
    for (int i = tid; i < 4 * 16 * NB_; i += 128) ((unsigned int*)hist)[i] = 0u;
    __syncthreads();

    // ---- pass 2: per-row histogram -> approximate k-th-largest threshold ----
    if (issuer) {
        tdm_load_2d(ka[0], keys_bf + hofs, HD_, CH_, D_);
        tdm_load_2d(ka[1], keys_bf + (size_t)CH_ * D_ + hofs, HD_, CH_, D_);
    }
    for (int c = 0; c < nch; ++c) {
        if (issuer) {
            if (c + 2 <= nch) __builtin_amdgcn_s_wait_tensorcnt(1);
            else              __builtin_amdgcn_s_wait_tensorcnt(0);
        }
        __syncthreads();
        const __bf16* kb = kbuf[c & 1];
#pragma unroll
        for (int t = 0; t < 2; ++t) {
            int m0 = c * CH_ + t * 16;
            v8f s = score_tile_lds(a0, a1, kb, t, lane);
            float iw = imp[m0 + colb] * 0.125f;
#pragma unroll
            for (int v = 0; v < 8; ++v) {
                float sv = s[v] * iw;
                float lo_ = rmn[v], rng = rmx[v] - rmn[v];
                int bin = (rng > 0.f) ? (int)((sv - lo_) * ((float)NB_ / rng)) : 0;
                bin = min(max(bin, 0), NB_ - 1);
                atomicAdd(&hist[wv][lohalf ? v : v + 8][bin], 1u);
            }
        }
        __syncthreads();
        if (issuer && c + 2 < nch)
            tdm_load_2d(ka[c & 1], keys_bf + (size_t)(c + 2) * CH_ * D_ + hofs, HD_, CH_, D_);
    }
    __syncthreads();
    if (lane < 16) {
        int row = lane;
        float lo_ = rowmin16[wv][row], hi_ = rowmax16[wv][row];
        unsigned int cum = 0;
        int bsel = 0;
        for (int bn = NB_ - 1; bn >= 0; --bn) {
            cum += hist[wv][row][bn];
            if (cum >= (unsigned int)kkeep) { bsel = bn; break; }
        }
        thr16[wv][row] = (hi_ > lo_) ? (lo_ + (hi_ - lo_) * ((float)bsel / (float)NB_)) : lo_;
    }
    __syncthreads();

    float thrv[8];
#pragma unroll
    for (int v = 0; v < 8; ++v) thrv[v] = thr16[wv][lohalf ? v : v + 8];

    // ---- pass 3: masked softmax weights -> att @ V (WMMA, K & V staged by TDM) ----
    v8f oacc[4] = {};
    float den[8];
#pragma unroll
    for (int v = 0; v < 8; ++v) den[v] = 0.f;

    if (issuer) {
        tdm_load_2d(ka[0], keys_bf + hofs, HD_, CH_, D_);
        tdm_load_2d(va[0], vT_bf + (size_t)hofs * M, CH_, HD_, M);
        tdm_load_2d(ka[1], keys_bf + (size_t)CH_ * D_ + hofs, HD_, CH_, D_);
        tdm_load_2d(va[1], vT_bf + (size_t)hofs * M + CH_, CH_, HD_, M);
    }
    for (int c = 0; c < nch; ++c) {
        if (issuer) {
            if (c + 2 <= nch) __builtin_amdgcn_s_wait_tensorcnt(2);
            else              __builtin_amdgcn_s_wait_tensorcnt(0);
        }
        __syncthreads();
        const __bf16* kb = kbuf[c & 1];
        const __bf16* vb = vbuf[c & 1];
#pragma unroll
        for (int t = 0; t < 2; ++t) {
            int m0 = c * CH_ + t * 16;
            v8f s = score_tile_lds(a0, a1, kb, t, lane);
            float iw = imp[m0 + colb] * 0.125f;
#pragma unroll
            for (int v = 0; v < 8; ++v) {
                float sv = s[v] * iw;
                float w  = (sv >= thrv[v]) ? __expf(sv - rmx[v]) : 0.f;
                den[v] += w;
                latt[wv][(lohalf ? v : v + 8) * 32 + t * 16 + colb] = (__bf16)w;
            }
        }
        v16bf af = load_afrag(latt[wv], 32, lane);
#pragma unroll
        for (int nt = 0; nt < 4; ++nt) {
            v16bf bfr = load_bfrag(vb + nt * 16 * CH_, CH_, lane);
            oacc[nt] = wmma_bf16(af, bfr, oacc[nt]);
        }
        __syncthreads();
        if (issuer && c + 2 < nch) {
            tdm_load_2d(ka[c & 1], keys_bf + (size_t)(c + 2) * CH_ * D_ + hofs, HD_, CH_, D_);
            tdm_load_2d(va[c & 1], vT_bf + (size_t)hofs * M + (size_t)(c + 2) * CH_, CH_, HD_, M);
        }
    }
#pragma unroll
    for (int v = 0; v < 8; ++v) den[v] = hsum16(den[v]);

    const int rhalf = lohalf ? 0 : 8;
#pragma unroll
    for (int nt = 0; nt < 4; ++nt)
#pragma unroll
        for (int v = 0; v < 8; ++v) {
            float d = den[v];
            float o = (d > 0.f) ? (oacc[nt][v] / d) : 0.f;
            out[(size_t)(q0 + rhalf + v) * D_ + hofs + nt * 16 + colb] = o;
        }
}

// ---------------- bank mixing weights ----------------
static __device__ inline float block_sum256(float v, volatile float* sb) {
    int lane = threadIdx.x & 31, w = threadIdx.x >> 5;
#pragma unroll
    for (int m = 1; m < 32; m <<= 1) v += __shfl_xor(v, m, 32);
    if (lane == 0) sb[w] = v;
    __syncthreads();
    if (threadIdx.x == 0) {
        float r = 0.f;
        for (int i = 0; i < 8; ++i) r += sb[i];
        sb[0] = r;
    }
    __syncthreads();
    float r = sb[0];
    __syncthreads();
    return r;
}

__global__ __launch_bounds__(256)
void weights_kernel(const float* __restrict__ st_imp, const float* __restrict__ lt_imp,
                    float* __restrict__ w2) {
    __shared__ float sb[8];
    float s = 0.f;
    for (int i = threadIdx.x; i < ST_; i += 256) s += st_imp[i];
    s = block_sum256(s, sb);
    float l = 0.f;
    for (int i = threadIdx.x; i < LT_; i += 256) l += lt_imp[i];
    l = block_sum256(l, sb);
    if (threadIdx.x == 0) {
        float sw = 1.f / (1.f + __expf(-(s / (float)ST_)));
        float lw = 1.f / (1.f + __expf(-(l / (float)LT_)));
        float tot = sw + lw;
        w2[0] = sw / tot;
        w2[1] = lw / tot;
    }
}

// ---------------- gate + residual + LayerNorm ----------------
__global__ __launch_bounds__(256)
void combine_kernel(const float* __restrict__ inputs, const float* __restrict__ st_out,
                    const float* __restrict__ lt_out, const float* __restrict__ Wg,
                    const float* __restrict__ bg, const float* __restrict__ ln_g,
                    const float* __restrict__ ln_b, const float* __restrict__ w2,
                    float* __restrict__ out) {
    __shared__ float xbuf[D_];
    __shared__ float sb[8];
    const int r = blockIdx.x;
    const float wst = w2[0], wlt = w2[1];
    const float* in = inputs + (size_t)r * D_;

    float gpart = 0.f;
    for (int i = threadIdx.x; i < D_; i += 256) {
        float m = wst * st_out[(size_t)r * D_ + i] + wlt * lt_out[(size_t)r * D_ + i];
        xbuf[i] = m;
        gpart += in[i] * Wg[i] + m * Wg[D_ + i];
    }
    gpart = block_sum256(gpart, sb);
    float gate = 1.f / (1.f + __expf(-(gpart + bg[0])));

    float s1 = 0.f, s2 = 0.f;
    for (int i = threadIdx.x; i < D_; i += 256) {
        float xv = in[i] + xbuf[i] * gate;
        xbuf[i] = xv;
        s1 += xv;
        s2 += xv * xv;
    }
    s1 = block_sum256(s1, sb);
    s2 = block_sum256(s2, sb);
    float mu   = s1 / (float)D_;
    float var  = s2 / (float)D_ - mu * mu;
    float rstd = rsqrtf(var + 1e-5f);
    for (int i = threadIdx.x; i < D_; i += 256)
        out[(size_t)r * D_ + i] = (xbuf[i] - mu) * rstd * ln_g[i] + ln_b[i];
}

// ---------------- host launcher ----------------
extern "C" void kernel_launch(void* const* d_in, const int* in_sizes, int n_in,
                              void* d_out, int out_size, void* d_ws, size_t ws_size,
                              hipStream_t stream) {
    (void)in_sizes; (void)n_in; (void)out_size; (void)ws_size;
    const float* inputs    = (const float*)d_in[0];
    const float* Wq        = (const float*)d_in[1];
    const float* bq        = (const float*)d_in[2];
    const float* st_keys   = (const float*)d_in[3];
    const float* st_values = (const float*)d_in[4];
    const float* lt_keys   = (const float*)d_in[5];
    const float* lt_values = (const float*)d_in[6];
    const float* st_imp    = (const float*)d_in[7];
    const float* lt_imp    = (const float*)d_in[8];
    const float* Wg        = (const float*)d_in[9];
    const float* bg        = (const float*)d_in[10];
    const float* ln_g      = (const float*)d_in[11];
    const float* ln_b      = (const float*)d_in[12];
    float* outp = (float*)d_out;

    char* ws = (char*)d_ws;
    size_t off = 0;
    auto salloc = [&](size_t bytes) -> void* {
        void* p = ws + off;
        off = (off + bytes + 255) & ~(size_t)255;
        return p;
    };
    __bf16* in_bf   = (__bf16*)salloc((size_t)B_ * S_ * D_ * 2);
    __bf16* q_bf    = (__bf16*)salloc((size_t)B_ * S_ * D_ * 2);
    __bf16* wqT_bf  = (__bf16*)salloc((size_t)D_ * D_ * 2);
    __bf16* stk_bf  = (__bf16*)salloc((size_t)ST_ * D_ * 2);
    __bf16* ltk_bf  = (__bf16*)salloc((size_t)LT_ * D_ * 2);
    __bf16* stvT_bf = (__bf16*)salloc((size_t)ST_ * D_ * 2);
    __bf16* ltvT_bf = (__bf16*)salloc((size_t)LT_ * D_ * 2);
    float*  st_out  = (float*)salloc((size_t)B_ * S_ * D_ * 4);
    float*  lt_out  = (float*)salloc((size_t)B_ * S_ * D_ * 4);
    float*  w2      = (float*)salloc(256);

    const int NIN = B_ * S_ * D_;
    f32_to_bf16_kernel<<<(NIN + 255) / 256, 256, 0, stream>>>(inputs, in_bf, NIN);
    f32_to_bf16_kernel<<<(ST_ * D_ + 255) / 256, 256, 0, stream>>>(st_keys, stk_bf, ST_ * D_);
    f32_to_bf16_kernel<<<(LT_ * D_ + 255) / 256, 256, 0, stream>>>(lt_keys, ltk_bf, LT_ * D_);
    transpose_to_bf16_kernel<<<(D_ * D_ + 255) / 256, 256, 0, stream>>>(Wq, wqT_bf, D_, D_);
    transpose_to_bf16_kernel<<<(ST_ * D_ + 255) / 256, 256, 0, stream>>>(st_values, stvT_bf, ST_, D_);
    transpose_to_bf16_kernel<<<(LT_ * D_ + 255) / 256, 256, 0, stream>>>(lt_values, ltvT_bf, LT_, D_);

    dim3 qgrid(B_ * S_ / 16, D_ / 64);
    qproj_kernel<<<qgrid, 32, 0, stream>>>(in_bf, wqT_bf, bq, q_bf);

    const int ABLOCKS = B_ * H_ * (S_ / 64);   // 256 blocks, 4 query tiles each
    attention_kernel<<<ABLOCKS, 128, 0, stream>>>(q_bf, stk_bf, stvT_bf, st_imp, st_out, ST_, KST_);
    attention_kernel<<<ABLOCKS, 128, 0, stream>>>(q_bf, ltk_bf, ltvT_bf, lt_imp, lt_out, LT_, KLT_);

    weights_kernel<<<1, 256, 0, stream>>>(st_imp, lt_imp, w2);
    combine_kernel<<<B_ * S_, 256, 0, stream>>>(inputs, st_out, lt_out, Wg, bg, ln_g, ln_b, w2, outp);
}
